// DCCF_22196390986323
// MI455X (gfx1250) — matrix operand order repaired and checked by reading, not compiled, not run on hardware
//
#include <hip/hip_runtime.h>
#include <hip/hip_bf16.h>

// ---------------- problem constants (match reference) ----------------
constexpr int N_USERS = 50000;
constexpr int N_ITEMS = 100000;
constexpr int EMB     = 64;
constexpr int NINT    = 128;
constexpr int NLAYERS = 2;
constexpr int NN      = N_USERS + N_ITEMS;     // 150000
constexpr int ND      = NN * EMB;              // 9,600,000 floats

typedef __attribute__((ext_vector_type(16))) __bf16 v16bf;
typedef __attribute__((ext_vector_type(8)))  float  v8f;

static __host__ __device__ __forceinline__ int cdiv(long long a, long long b) {
    return (int)((a + b - 1) / b);
}

// Contiguous k-order mapping this file uses for 16-bit WMMA A-frags (not the ISA table order):
// k-local kl in [0,32) -> (halfd = kl>>4, j).  Lanes 0-15 hold kl 0-15;
// lanes 16-31 hold kl 16-31.  Element j within the per-lane v16bf.
__device__ __forceinline__ void a_fragpos(int kl, int& halfd, int& j) {
    halfd = kl >> 4;
    j = kl & 15;
}

// ---------------- elementwise / setup kernels ----------------
__global__ void k_zero(float* __restrict__ p, int n) {
    int i = blockIdx.x * blockDim.x + threadIdx.x;
    if (i < n) p[i] = 0.f;
}

__global__ void k_init_emb(const float* __restrict__ ue, const float* __restrict__ ie,
                           float* __restrict__ emb, float* __restrict__ acc) {
    int i = blockIdx.x * blockDim.x + threadIdx.x;
    if (i < ND) {
        float v = (i < N_USERS * EMB) ? ue[i] : ie[i - N_USERS * EMB];
        emb[i] = v;
        acc[i] = v;
    }
}

__global__ void k_deg(const int* __restrict__ h, float* __restrict__ deg, int E) {
    int e = blockIdx.x * blockDim.x + threadIdx.x;
    if (e < E) atomicAdd(&deg[h[e]], 1.0f);
}

__global__ void k_dinv(const float* __restrict__ deg, float* __restrict__ dinv, int n) {
    int i = blockIdx.x * blockDim.x + threadIdx.x;
    if (i < n) {
        float d = deg[i];
        dinv[i] = (d > 0.f) ? rsqrtf(d) : 0.f;
    }
}

__global__ void k_gvals(const int* __restrict__ h, const int* __restrict__ t,
                        const float* __restrict__ dinv, float* __restrict__ g, int E) {
    int e = blockIdx.x * blockDim.x + threadIdx.x;
    if (e < E) g[e] = dinv[h[e]] * dinv[t[e]];
}

// ---------------- SpMM: y[h[e]] += val_e * x[t[e]] (scatter-add) ----------------
__global__ void k_spmm(const int* __restrict__ h, const int* __restrict__ t,
                       const float* __restrict__ val, const float* __restrict__ x,
                       float* __restrict__ y, int E) {
    long long gid = (long long)blockIdx.x * blockDim.x + threadIdx.x;
    if (gid >= (long long)E * 16) return;
    int e  = (int)(gid >> 4);
    int d4 = ((int)gid & 15) * 4;
    float v = val[e];
    const float4 xv = *(const float4*)(x + (long long)t[e] * EMB + d4);
    float* dst = y + (long long)h[e] * EMB + d4;
    atomicAdd(dst + 0, v * xv.x);
    atomicAdd(dst + 1, v * xv.y);
    atomicAdd(dst + 2, v * xv.z);
    atomicAdd(dst + 3, v * xv.w);
}

__global__ void k_spmm_adapt(const int* __restrict__ h, const int* __restrict__ t,
                             const float* __restrict__ alpha, const float* __restrict__ rowsum,
                             const float* __restrict__ x, float* __restrict__ y, int E) {
    long long gid = (long long)blockIdx.x * blockDim.x + threadIdx.x;
    if (gid >= (long long)E * 16) return;
    int e  = (int)(gid >> 4);
    int d4 = ((int)gid & 15) * 4;
    int he = h[e];
    float rs = rowsum[he];
    float v = (rs > 0.f) ? alpha[e] / rs : 0.f;
    const float4 xv = *(const float4*)(x + (long long)t[e] * EMB + d4);
    float* dst = y + (long long)he * EMB + d4;
    atomicAdd(dst + 0, v * xv.x);
    atomicAdd(dst + 1, v * xv.y);
    atomicAdd(dst + 2, v * xv.z);
    atomicAdd(dst + 3, v * xv.w);
}

// ---------------- row inverse L2 norms ----------------
__global__ void k_invnorm(const float* __restrict__ x, float* __restrict__ invn, int n) {
    long long gid = (long long)blockIdx.x * blockDim.x + threadIdx.x;
    if (gid >= (long long)n * 8) return;
    int r = (int)(gid >> 3), l = (int)gid & 7;
    const float* row = x + (long long)r * EMB + l * 8;
    float s = 0.f;
    #pragma unroll
    for (int c = 0; c < 8; ++c) s += row[c] * row[c];
    #pragma unroll
    for (int m = 4; m; m >>= 1) s += __shfl_xor(s, m, 8);
    if (l == 0) invn[r] = 1.f / fmaxf(sqrtf(s), 1e-8f);
}

// ---------------- per-edge adaptive alpha + rowsum ----------------
__global__ void k_alpha(const int* __restrict__ h, const int* __restrict__ t,
                        const float* __restrict__ x, const float* __restrict__ invn,
                        float* __restrict__ alpha, float* __restrict__ rowsum, int E) {
    long long gid = (long long)blockIdx.x * blockDim.x + threadIdx.x;
    if (gid >= (long long)E * 8) return;
    int e = (int)(gid >> 3), l = (int)gid & 7;
    int he = h[e], te = t[e];
    const float* ph = x + (long long)he * EMB + l * 8;
    const float* pt = x + (long long)te * EMB + l * 8;
    float dot = 0.f;
    #pragma unroll
    for (int c = 0; c < 8; ++c) dot += ph[c] * pt[c];
    #pragma unroll
    for (int m = 4; m; m >>= 1) dot += __shfl_xor(dot, m, 8);
    if (l == 0) {
        float a = (dot * invn[he] * invn[te] + 1.f) * 0.5f;
        alpha[e] = a;
        atomicAdd(&rowsum[he], a);
    }
}

// ---------------- layer update (float4): emb += sum ; acc += emb ----------------
__global__ void k_update4(const float4* __restrict__ gnn, const float4* __restrict__ inte,
                          const float4* __restrict__ gaa, const float4* __restrict__ iaa,
                          float4* __restrict__ emb, float4* __restrict__ acc, int n4) {
    int i = blockIdx.x * blockDim.x + threadIdx.x;
    if (i < n4) {
        float4 g = gnn[i], s = inte[i], ga = gaa[i], ia = iaa[i], e = emb[i], a = acc[i];
        e.x += g.x + s.x + ga.x + ia.x;
        e.y += g.y + s.y + ga.y + ia.y;
        e.z += g.z + s.z + ga.z + ia.z;
        e.w += g.w + s.w + ga.w + ia.w;
        a.x += e.x; a.y += e.y; a.z += e.z; a.w += e.w;
        emb[i] = e;
        acc[i] = a;
    }
}

// ---------------- intent transform: Y = softmax(X @ W) @ W^T  (WMMA bf16) ----------------
// Block = 64 threads = 2 waves; each wave owns a 16-row tile (32 rows / block).
// All WMMA operands are staged in contiguous-k fragment order so per-lane fragment loads are
// contiguous 32B vectors (ds_load_b128 x2) instead of 16 scalar u16 loads + packs.
__global__ __launch_bounds__(64) void k_intent(const float* __restrict__ x,
                                               const float* __restrict__ W,
                                               float* __restrict__ y, int nrows) {
    __shared__ __bf16 sWt[NINT * EMB];       // W^T: [n][k], B-frags for GEMM1 (16 KB)
    __shared__ __bf16 sWo[EMB * NINT];       // W  : [d][k], B-frags for GEMM2 (16 KB)
    __shared__ v16bf  sXfrag[2][2][32];      // [wave][kstep][lane] A-frags of X  (4 KB)
    __shared__ v16bf  sSfrag[2][4][32];      // [wave][kstep][lane] A-frags of S  (8 KB)

    const int tid = threadIdx.x;

    // stage W in both orientations as bf16
    for (int i = tid; i < EMB * NINT; i += 64) {
        int k = i >> 7, n = i & 127;          // W[k][n]
        __bf16 v = (__bf16)W[i];
        sWo[i] = v;                           // row-major W  -> contiguous in k per column d
        sWt[n * EMB + k] = v;                 // W^T          -> contiguous in k per intent n
    }
    __syncthreads();

    const int w    = tid >> 5;
    const int lane = tid & 31;
    const int half = lane >> 4;
    const int m16  = lane & 15;
    const int row0 = (blockIdx.x * 2 + w) * 16;
    if (row0 >= nrows) return;   // wave-uniform: EXEC all-ones for WMMA

    // prefetch next tile's X rows (global_prefetch_b8; speculative-safe)
    __builtin_prefetch(x + (long long)(row0 + 32) * EMB + lane * 8, 0, 1);

    // stage X tile (16x64) directly into A-fragment order, float4 global loads
    for (int idx = lane; idx < 16 * 16; idx += 32) {
        int r = idx >> 4, c4 = (idx & 15) * 4;
        int gr = row0 + r;
        float4 xv = make_float4(0.f, 0.f, 0.f, 0.f);
        if (gr < nrows) xv = *(const float4*)(x + (long long)gr * EMB + c4);
        float xa[4] = {xv.x, xv.y, xv.z, xv.w};
        #pragma unroll
        for (int u = 0; u < 4; ++u) {
            int c = c4 + u;
            int ks = c >> 5, kl = c & 31, hd, j;
            a_fragpos(kl, hd, j);
            ((__bf16*)&sXfrag[w][ks][r + 16 * hd])[j] = (__bf16)xa[u];
        }
    }
    asm volatile("" ::: "memory");

    // ---- GEMM1: Z(16x128) = X(16x64) @ W(64x128); keep all 8 accumulators live ----
    v8f accs[8];
    #pragma unroll
    for (int n0 = 0; n0 < 8; ++n0) {
        v8f acc = {};
        #pragma unroll
        for (int kq = 0; kq < 2; ++kq) {          // ks = 32*kq
            v16bf a = sXfrag[w][kq][lane];
            v16bf b = *reinterpret_cast<const v16bf*>(
                &sWt[(n0 * 16 + m16) * EMB + 32 * kq + 16 * half]);
            acc = __builtin_amdgcn_wmma_f32_16x16x32_bf16(
                false, a, false, b, (short)0, acc, false, false);
        }
        accs[n0] = acc;
    }

    // ---- softmax across 128 intents per row, entirely in registers + shuffles ----
    // element i of each acc holds row r = i + 8*half, col n0*16 + m16.
    // shfl_xor masks 1,2,4,8 flip only the n-lane bits -> reduction stays in half-group.
    #pragma unroll
    for (int i = 0; i < 8; ++i) {
        float mx = -3.4e38f;
        #pragma unroll
        for (int n0 = 0; n0 < 8; ++n0) mx = fmaxf(mx, accs[n0][i]);
        #pragma unroll
        for (int m = 1; m <= 8; m <<= 1) mx = fmaxf(mx, __shfl_xor(mx, m, 32));
        float sum = 0.f;
        #pragma unroll
        for (int n0 = 0; n0 < 8; ++n0) {
            float ev = __expf(accs[n0][i] - mx);
            accs[n0][i] = ev;
            sum += ev;
        }
        #pragma unroll
        for (int m = 1; m <= 8; m <<= 1) sum += __shfl_xor(sum, m, 32);
        float inv = 1.f / sum;
        #pragma unroll
        for (int n0 = 0; n0 < 8; ++n0) accs[n0][i] *= inv;
    }

    // scatter probabilities into A-fragment order for GEMM2 (bf16)
    #pragma unroll
    for (int n0 = 0; n0 < 8; ++n0) {
        int k  = n0 * 16 + m16;
        int ks = k >> 5, kl = k & 31, hd, j;
        a_fragpos(kl, hd, j);
        #pragma unroll
        for (int i = 0; i < 8; ++i) {
            int r = i + 8 * half;
            ((__bf16*)&sSfrag[w][ks][r + 16 * hd])[j] = (__bf16)accs[n0][i];
        }
    }
    asm volatile("" ::: "memory");

    // ---- GEMM2: Y(16x64) = S(16x128) @ W^T(128x64) ----
    #pragma unroll
    for (int d0 = 0; d0 < 4; ++d0) {
        v8f acc = {};
        #pragma unroll
        for (int kq = 0; kq < 4; ++kq) {          // ks = 32*kq over intents
            v16bf a = sSfrag[w][kq][lane];
            v16bf b = *reinterpret_cast<const v16bf*>(
                &sWo[(d0 * 16 + m16) * NINT + 32 * kq + 16 * half]);  // W^T[k][d] = W[d][k]
            acc = __builtin_amdgcn_wmma_f32_16x16x32_bf16(
                false, a, false, b, (short)0, acc, false, false);
        }
        #pragma unroll
        for (int i = 0; i < 8; ++i) {
            int gr = row0 + i + 8 * half;
            if (gr < nrows) y[(long long)gr * EMB + d0 * 16 + m16] = acc[i];
        }
    }
}

// ---------------- launcher ----------------
extern "C" void kernel_launch(void* const* d_in, const int* in_sizes, int n_in,
                              void* d_out, int out_size, void* d_ws, size_t ws_size,
                              hipStream_t stream) {
    const float* ue = (const float*)d_in[0];
    const float* ie = (const float*)d_in[1];
    const float* uW = (const float*)d_in[2];
    const float* iW = (const float*)d_in[3];
    const int*   h  = (const int*)d_in[4];
    const int*   t  = (const int*)d_in[5];
    const int    E  = in_sizes[4];
    float* out = (float*)d_out;

    // workspace layout (floats)
    float* ws     = (float*)d_ws;
    float* emb    = ws;
    float* gnn    = emb  + ND;
    float* inte   = gnn  + ND;
    float* gaa    = inte + ND;
    float* iaa    = gaa  + ND;
    float* gvals  = iaa  + ND;      // E
    float* alpha  = gvals + E;      // E
    float* deg    = alpha + E;      // NN
    float* dinv   = deg   + NN;     // NN
    float* rowsum = dinv  + NN;     // NN
    float* invn   = rowsum + NN;    // NN

    const int B = 256;

    k_init_emb<<<cdiv(ND, B), B, 0, stream>>>(ue, ie, emb, out);

    k_zero<<<cdiv(NN, B), B, 0, stream>>>(deg, NN);
    k_deg<<<cdiv(E, B), B, 0, stream>>>(h, deg, E);
    k_dinv<<<cdiv(NN, B), B, 0, stream>>>(deg, dinv, NN);
    k_gvals<<<cdiv(E, B), B, 0, stream>>>(h, t, dinv, gvals, E);

    for (int layer = 0; layer < NLAYERS; ++layer) {
        // gnn = spmm(g_values, emb)
        k_zero<<<cdiv(ND, B), B, 0, stream>>>(gnn, ND);
        k_spmm<<<cdiv((long long)E * 16, B), B, 0, stream>>>(h, t, gvals, emb, gnn, E);

        // intent path (WMMA): users then items
        k_intent<<<cdiv(N_USERS, 32), 64, 0, stream>>>(emb, uW, inte, N_USERS);
        k_intent<<<cdiv(N_ITEMS, 32), 64, 0, stream>>>(emb + (long long)N_USERS * EMB, iW,
                                                       inte + (long long)N_USERS * EMB, N_ITEMS);

        // gaa = spmm(adaptive(gnn), emb)
        k_invnorm<<<cdiv((long long)NN * 8, B), B, 0, stream>>>(gnn, invn, NN);
        k_zero<<<cdiv(NN, B), B, 0, stream>>>(rowsum, NN);
        k_alpha<<<cdiv((long long)E * 8, B), B, 0, stream>>>(h, t, gnn, invn, alpha, rowsum, E);
        k_zero<<<cdiv(ND, B), B, 0, stream>>>(gaa, ND);
        k_spmm_adapt<<<cdiv((long long)E * 16, B), B, 0, stream>>>(h, t, alpha, rowsum, emb, gaa, E);

        // iaa = spmm(adaptive(int_e), emb)
        k_invnorm<<<cdiv((long long)NN * 8, B), B, 0, stream>>>(inte, invn, NN);
        k_zero<<<cdiv(NN, B), B, 0, stream>>>(rowsum, NN);
        k_alpha<<<cdiv((long long)E * 8, B), B, 0, stream>>>(h, t, inte, invn, alpha, rowsum, E);
        k_zero<<<cdiv(ND, B), B, 0, stream>>>(iaa, ND);
        k_spmm_adapt<<<cdiv((long long)E * 16, B), B, 0, stream>>>(h, t, alpha, rowsum, emb, iaa, E);

        // emb = gnn + int_e + gaa + iaa + emb ; acc += emb
        k_update4<<<cdiv(ND / 4, B), B, 0, stream>>>((const float4*)gnn, (const float4*)inte,
                                                     (const float4*)gaa, (const float4*)iaa,
                                                     (float4*)emb, (float4*)out, ND / 4);
    }
}
